// StarEEncoder_52467320487979
// MI455X (gfx1250) — compile-verified, matrix operand order checked by reading, not run on hardware
//
#include <hip/hip_runtime.h>
#include <hip/hip_bf16.h>

#define DIM   200
#define HDIM  100
#define KP    224          // K padded to 7*32
#define NP    208          // N padded to 13*16
#define NT    13
#define KS    7
#define NWAVES 4
#define TM    16
#define ROWS_PER_BLOCK (NWAVES*TM)   // 64
#define ALPHA_C 0.8f
#define OMALPHA 0.2f
#define BN_EPS_C 1e-5f

typedef __attribute__((ext_vector_type(16))) _Float16 v16h;
typedef __attribute__((ext_vector_type(8)))  float    v8f;

struct __align__(16) TileLDS {
  _Float16 wT[NP*KP];           // transposed weight: wT[n*KP+k] = w[k][n], f16, zero padded
  _Float16 A[NWAVES][TM*KP];    // per-wave rotated A rows (16 x KP), f16, zero padded
  int   idx[NWAVES][TM];        // scatter destination row per A-row
  float scl[NWAVES][TM];        // per-row scale (0 => skip scatter)
};

// Pack a 16xf16 WMMA operand fragment (A-row or B-column view) per ISA 7.12.2:
// lane holds K = kbase+{0,2,4,6} pairs and K = kbase+16+{0,2,4,6} pairs.
__device__ __forceinline__ v16h load_frag16(const _Float16* __restrict__ row, int kbase) {
  union { v16h v; unsigned u[8]; } f;
  const unsigned* p = (const unsigned*)row;   // 2 f16 per dword; kbase is even
  int b = kbase >> 1;
#pragma unroll
  for (int i = 0; i < 4; ++i) f.u[i] = p[b + i];
#pragma unroll
  for (int i = 0; i < 4; ++i) f.u[4 + i] = p[b + 8 + i];
  return f.v;
}

__device__ __forceinline__ void stage_weight(TileLDS& s, const float* __restrict__ w,
                                             int tid, int nthreads) {
  for (int t = tid; t < NP * KP; t += nthreads) {
    int n = t % NP;            // n fast -> coalesced global reads of w[k][n]
    int k = t / NP;
    float v = (n < DIM && k < DIM) ? w[k * DIM + n] : 0.0f;
    s.wT[n * KP + k] = (_Float16)v;
  }
}

// 16xDIM (A, in LDS) times DIMxDIM (wT, in LDS) -> scatter rows to dst.
__device__ __forceinline__ void wave_gemm_scatter(const TileLDS& s, int wave, int lane,
                                                  float* __restrict__ dst, bool useAtomic) {
  const _Float16* At = &s.A[wave][0];
  const int kb_lane = (lane < 16) ? 0 : 8;
  const int m_in = lane & 15;
  v16h afr[KS];
#pragma unroll
  for (int ks = 0; ks < KS; ++ks)
    afr[ks] = load_frag16(At + m_in * KP, ks * 32 + kb_lane);

  // hoist per-row scatter metadata out of the n-tile loop (invariant across nt)
  const int mbase = (lane < 16) ? 0 : 8;
  float scv[8];
  int   off[8];
#pragma unroll
  for (int r = 0; r < 8; ++r) {
    scv[r] = s.scl[wave][mbase + r];
    off[r] = s.idx[wave][mbase + r] * DIM;
  }

  for (int nt = 0; nt < NT; ++nt) {
    int n = nt * 16 + (lane & 15);
    v8f acc = {};
#pragma unroll
    for (int ks = 0; ks < KS; ++ks) {
      v16h b = load_frag16(&s.wT[n * KP], ks * 32 + kb_lane);
      acc = __builtin_amdgcn_wmma_f32_16x16x32_f16(false, afr[ks], false, b,
                                                   (short)0, acc, false, false);
    }
    if (n < DIM) {
#pragma unroll
      for (int r = 0; r < 8; ++r) {
        if (scv[r] != 0.0f) {
          float v = acc[r] * scv[r];
          if (useAtomic) atomicAdd(&dst[off[r] + n], v);
          else           dst[off[r] + n] = v;
        }
      }
    }
  }
}

__device__ __forceinline__ void zero_row(TileLDS& s, int wave, int m, int lane) {
  for (int d = lane; d < DIM; d += 32) s.A[wave][m * KP + d] = (_Float16)0.0f;
}

// ---------------- kernels ----------------

__global__ void deg_kernel(const int* __restrict__ row, float* __restrict__ deg, int ne) {
  int i = blockIdx.x * blockDim.x + threadIdx.x;
  if (i < ne) atomicAdd(&deg[row[i]], 1.0f);
}

__global__ void __launch_bounds__(128)
qual_kernel(const float* __restrict__ x, const float* __restrict__ rel,
            const int* __restrict__ qrel, const int* __restrict__ qent,
            const int* __restrict__ qidx, const float* __restrict__ wq,
            float* __restrict__ qw, int nq) {
  __shared__ TileLDS s;
  int tid = threadIdx.x, lane = tid & 31, wave = tid >> 5;
  stage_weight(s, wq, tid, blockDim.x);
  int base = (blockIdx.x * NWAVES + wave) * TM;
  for (int m = 0; m < TM; ++m) {
    int g = base + m;
    if (g < nq) {
      const float* h = x + (size_t)qent[g] * DIM;
      const float* r = rel + (size_t)qrel[g] * DIM;
      for (int d = lane; d < HDIM; d += 32) {
        float hre = h[d], him = h[d + HDIM], rre = r[d], rim = r[d + HDIM];
        s.A[wave][m * KP + d]        = (_Float16)(hre * rre - him * rim);
        s.A[wave][m * KP + d + HDIM] = (_Float16)(hre * rim + him * rre);
      }
      if (lane == 0) { s.idx[wave][m] = qidx[g]; s.scl[wave][m] = 1.0f; }
    } else {
      zero_row(s, wave, m, lane);
      if (lane == 0) { s.idx[wave][m] = 0; s.scl[wave][m] = 0.0f; }
    }
    for (int k = DIM + lane; k < KP; k += 32) s.A[wave][m * KP + k] = (_Float16)0.0f;
  }
  __syncthreads();
  wave_gemm_scatter(s, wave, lane, qw, true);
}

__global__ void __launch_bounds__(128)
edge_kernel(const float* __restrict__ x, const float* __restrict__ rel,
            const int* __restrict__ erow, const int* __restrict__ ecol,
            const int* __restrict__ etyp, const float* __restrict__ qw,
            const float* __restrict__ w, const float* __restrict__ deg,
            float* __restrict__ acc, int ne) {
  __shared__ TileLDS s;
  int tid = threadIdx.x, lane = tid & 31, wave = tid >> 5;
  stage_weight(s, w, tid, blockDim.x);
  int base = (blockIdx.x * NWAVES + wave) * TM;
  for (int m = 0; m < TM; ++m) {
    int g = base + m;
    if (g < ne) {
      int cl = ecol[g];
      int et = etyp[g];
      const float* h = x + (size_t)cl * DIM;
      const float* r = rel + (size_t)et * DIM;
      const float* q = qw + (size_t)g * DIM;
      for (int d = lane; d < HDIM; d += 32) {
        float hre = h[d], him = h[d + HDIM];
        float rre = ALPHA_C * r[d]        + OMALPHA * q[d];
        float rim = ALPHA_C * r[d + HDIM] + OMALPHA * q[d + HDIM];
        s.A[wave][m * KP + d]        = (_Float16)(hre * rre - him * rim);
        s.A[wave][m * KP + d + HDIM] = (_Float16)(hre * rim + him * rre);
      }
      if (lane == 0) {
        int rw = erow[g];
        float dr = deg[rw], dc = deg[cl];
        float di = dr > 0.0f ? 1.0f / sqrtf(dr) : 0.0f;
        float dj = dc > 0.0f ? 1.0f / sqrtf(dc) : 0.0f;
        s.idx[wave][m] = rw;
        s.scl[wave][m] = di * dj;
      }
    } else {
      zero_row(s, wave, m, lane);
      if (lane == 0) { s.idx[wave][m] = 0; s.scl[wave][m] = 0.0f; }
    }
    for (int k = DIM + lane; k < KP; k += 32) s.A[wave][m * KP + k] = (_Float16)0.0f;
  }
  __syncthreads();
  wave_gemm_scatter(s, wave, lane, acc, true);
}

__global__ void __launch_bounds__(128)
loop_kernel(const float* __restrict__ x, const float* __restrict__ lrel,
            const float* __restrict__ wl, float* __restrict__ acc, int nent) {
  __shared__ TileLDS s;
  int tid = threadIdx.x, lane = tid & 31, wave = tid >> 5;
  stage_weight(s, wl, tid, blockDim.x);
  int base = (blockIdx.x * NWAVES + wave) * TM;
  for (int m = 0; m < TM; ++m) {
    int g = base + m;
    if (g < nent) {
      const float* h = x + (size_t)g * DIM;
      for (int d = lane; d < HDIM; d += 32) {
        float hre = h[d], him = h[d + HDIM], rre = lrel[d], rim = lrel[d + HDIM];
        s.A[wave][m * KP + d]        = (_Float16)(hre * rre - him * rim);
        s.A[wave][m * KP + d + HDIM] = (_Float16)(hre * rim + him * rre);
      }
      if (lane == 0) { s.idx[wave][m] = g; s.scl[wave][m] = 1.0f; }
    } else {
      zero_row(s, wave, m, lane);
      if (lane == 0) { s.idx[wave][m] = 0; s.scl[wave][m] = 0.0f; }
    }
    for (int k = DIM + lane; k < KP; k += 32) s.A[wave][m * KP + k] = (_Float16)0.0f;
  }
  __syncthreads();
  wave_gemm_scatter(s, wave, lane, acc, false);   // plain stores: initializes accumulator
}

__global__ void stats_kernel(const float* __restrict__ acc, float* __restrict__ ssum,
                             float* __restrict__ ssq, int nrows) {
  int c = threadIdx.x;
  if (c >= DIM) return;
  int r0 = blockIdx.x * 256;
  int r1 = r0 + 256; if (r1 > nrows) r1 = nrows;
  float s = 0.0f, q = 0.0f;
  for (int r = r0; r < r1; ++r) {
    float v = acc[(size_t)r * DIM + c];
    s += v; q += v * v;
  }
  atomicAdd(&ssum[c], s);
  atomicAdd(&ssq[c], q);
}

__global__ void apply_kernel(float* __restrict__ acc, const float* __restrict__ ssum,
                             const float* __restrict__ ssq, const float* __restrict__ gamma,
                             const float* __restrict__ beta, int nrows) {
  long i = (long)blockIdx.x * blockDim.x + threadIdx.x;
  long total = (long)nrows * DIM;
  if (i >= total) return;
  int c = (int)(i % DIM);
  const float third = 1.0f / 3.0f;
  float invN = 1.0f / (float)nrows;
  float mean = ssum[c] * invN * third;
  float ex2  = ssq[c] * invN * third * third;
  float var  = ex2 - mean * mean;
  float v = acc[i] * third;
  float y = (v - mean) / sqrtf(var + BN_EPS_C) * gamma[c] + beta[c];
  acc[i] = tanhf(y);
}

__global__ void relout_kernel(const float* __restrict__ rel, const float* __restrict__ wr,
                              float* __restrict__ out, int nrel) {
  __shared__ float srow[DIM];
  int i = blockIdx.x;
  if (i >= nrel) return;
  for (int t = threadIdx.x; t < DIM; t += blockDim.x) srow[t] = rel[(size_t)i * DIM + t];
  __syncthreads();
  int n = threadIdx.x;
  if (n < DIM) {
    float s = 0.0f;
    for (int k = 0; k < DIM; ++k) s += srow[k] * wr[k * DIM + n];
    out[(size_t)i * DIM + n] = s;
  }
}

extern "C" void kernel_launch(void* const* d_in, const int* in_sizes, int n_in,
                              void* d_out, int out_size, void* d_ws, size_t ws_size,
                              hipStream_t stream) {
  const float* x        = (const float*)d_in[0];
  const int*   eidx     = (const int*)d_in[1];
  const int*   etyp     = (const int*)d_in[2];
  const float* rel      = (const float*)d_in[3];
  const int*   quals    = (const int*)d_in[4];
  const float* w_in     = (const float*)d_in[5];
  const float* w_out    = (const float*)d_in[6];
  const float* w_loop   = (const float*)d_in[7];
  const float* w_rel    = (const float*)d_in[8];
  const float* w_q      = (const float*)d_in[9];
  const float* loop_rel = (const float*)d_in[10];
  const float* gamma    = (const float*)d_in[11];
  const float* beta     = (const float*)d_in[12];

  const int NENT = in_sizes[0] / DIM;   // 50000
  const int E2   = in_sizes[2];         // 400000
  const int E    = E2 / 2;              // 200000
  const int Q2   = in_sizes[4] / 3;     // 200000
  const int Q    = Q2 / 2;              // 100000
  const int NREL = in_sizes[3] / DIM;   // 400

  float* acc    = (float*)d_out;                     // [NENT][DIM], pre-BN accumulator then final
  float* relout = acc + (size_t)NENT * DIM;          // [NREL][DIM]

  float* qw   = (float*)d_ws;                        // [E][DIM]
  float* deg  = qw + (size_t)E * DIM;                // [NENT]
  float* ssum = deg + NENT;                          // [DIM]
  float* ssq  = ssum + DIM;                          // [DIM]

  // 1) loop result initializes the accumulator (plain stores cover every element)
  loop_kernel<<<(NENT + ROWS_PER_BLOCK - 1) / ROWS_PER_BLOCK, 128, 0, stream>>>(
      x, loop_rel, w_loop, acc, NENT);

  // 2) both directions: deg -> qualifier buffer -> edge messages
  for (int dir = 0; dir < 2; ++dir) {
    const int* erow = eidx + (size_t)dir * E;
    const int* ecol = eidx + (size_t)E2 + (size_t)dir * E;
    const int* et   = etyp + (size_t)dir * E;
    const int* qr   = quals + (size_t)dir * Q;
    const int* qe   = quals + (size_t)Q2 + (size_t)dir * Q;
    const int* qi   = quals + (size_t)2 * Q2 + (size_t)dir * Q;
    const float* w  = dir ? w_out : w_in;

    hipMemsetAsync(qw, 0, (size_t)E * DIM * sizeof(float), stream);
    hipMemsetAsync(deg, 0, (size_t)NENT * sizeof(float), stream);
    deg_kernel<<<(E + 255) / 256, 256, 0, stream>>>(erow, deg, E);
    qual_kernel<<<(Q + ROWS_PER_BLOCK - 1) / ROWS_PER_BLOCK, 128, 0, stream>>>(
        x, rel, qr, qe, qi, w_q, qw, Q);
    edge_kernel<<<(E + ROWS_PER_BLOCK - 1) / ROWS_PER_BLOCK, 128, 0, stream>>>(
        x, rel, erow, ecol, et, qw, w, deg, acc, E);
  }

  // 3) batch-norm + tanh
  hipMemsetAsync(ssum, 0, 2 * DIM * sizeof(float), stream);
  stats_kernel<<<(NENT + 255) / 256, 256, 0, stream>>>(acc, ssum, ssq, NENT);
  long total = (long)NENT * DIM;
  apply_kernel<<<(int)((total + 255) / 256), 256, 0, stream>>>(acc, ssum, ssq, gamma, beta, NENT);

  // 4) relation output (exact f32, tiny)
  relout_kernel<<<NREL, 256, 0, stream>>>(rel, w_rel, relout, NREL);
}
